// SVDpp_5703716569706
// MI455X (gfx1250) — matrix operand order repaired and checked by reading, not compiled, not run on hardware
//
#include <hip/hip_runtime.h>
#include <hip/hip_bf16.h>

typedef __attribute__((ext_vector_type(16))) _Float16 v16h;
typedef __attribute__((ext_vector_type(8)))  _Float16 v8h;
typedef __attribute__((ext_vector_type(8)))  float    v8f;

#define DIM 64
#define U_MEAN 0.05f

// ---------------------------------------------------------------------------
// Kernel 1: convert Q1 (f32, [I,64]) -> f16 copy in workspace for WMMA B-matrix
// ---------------------------------------------------------------------------
__global__ void q1_to_f16_kernel(const float* __restrict__ q1,
                                 _Float16* __restrict__ q1h, int n) {
    int i = blockIdx.x * blockDim.x + threadIdx.x;
    if (i < n) q1h[i] = (_Float16)q1[i];
}

// ---------------------------------------------------------------------------
// Kernel 2: per-user ragged segment reduce.
//   One 64-thread block per user; seg_ids is sorted so the segment bounds are
//   found by binary search (lower_bound(u), lower_bound(u+1)). Items are
//   staged through LDS in chunks of 64; thread t owns dim t.
//   P_full[u,d] = Ysum[u,d]*rsqrt(count[u]) + P[u,d], emitted as f16 (A-matrix).
// ---------------------------------------------------------------------------
__global__ void pfull_kernel(const float* __restrict__ Q2,
                             const float* __restrict__ P,
                             const int* __restrict__ flat_items,
                             const int* __restrict__ seg_ids,
                             _Float16* __restrict__ Ph,
                             int nnz) {
    __shared__ int s_range[2];
    __shared__ int s_items[64];
    const int u = blockIdx.x;
    const int t = threadIdx.x;            // 0..63, dim index

    if (t < 2) {
        int key = u + t;                  // lower_bound(seg_ids, key)
        int lo = 0, hi = nnz;
        while (lo < hi) {
            int mid = (lo + hi) >> 1;
            if (seg_ids[mid] < key) lo = mid + 1; else hi = mid;
        }
        s_range[t] = lo;
    }
    __syncthreads();
    const int start = s_range[0];
    const int end   = s_range[1];

    float acc = 0.0f;
    for (int j0 = start; j0 < end; j0 += 64) {
        int chunk = min(64, end - j0);
        if (t < chunk) s_items[t] = flat_items[j0 + t];
        __syncthreads();
        for (int k = 0; k < chunk; ++k)
            acc += Q2[s_items[k] * DIM + t];   // 2MB table: stays hot in L2
        __syncthreads();
    }

    float cnt = (float)(end - start);     // every user has >= 1 item
    float val = acc * rsqrtf(cnt) + P[u * DIM + t];
    Ph[u * DIM + t] = (_Float16)val;
}

// ---------------------------------------------------------------------------
// Kernel 3: WMMA GEMM + fused bias/sigmoid epilogue.
//   out[u,i] = 5*sigmoid(dot(P_full[u,:], Q1[i,:]) + bu[u] + bi[i] + U_MEAN)
//   One wave computes a 16x64 strip: A fragment reused across 4 N-tiles,
//   K=64 via two v_wmma_f32_16x16x32_f16 per tile. Output stores are
//   non-temporal: 512 MB written once, never re-read -> rinse to HBM,
//   keep L2 for the f16 operand tables.
// ---------------------------------------------------------------------------
__global__ __launch_bounds__(256) void gemm_sigmoid_kernel(
        const _Float16* __restrict__ Ph,   // [U,64] row-major f16 (A)
        const _Float16* __restrict__ Q1h,  // [I,64] row-major f16 (B, transposed use)
        const float* __restrict__ bu,
        const float* __restrict__ bi,
        float* __restrict__ out,           // [U,I] f32
        int n_items) {
    const int lane = threadIdx.x & 31;
    const int wave = threadIdx.x >> 5;
    const int half = lane >> 4;            // which 16-lane half of the wave
    const int l16  = lane & 15;

    const int u_base  = (blockIdx.y * 8 + wave) * 16;  // M tile (16 rows)
    const int i_base0 = blockIdx.x * 64;               // 4 N tiles (64 cols)

    // ---- A fragments (16x32, contiguous K per half-wave, same order as B) ----
    // lane (M = l16): half 0 owns K {k0+0..15}, half 1 owns
    // K {k0+16..31} (matches B); 2 f16 per VGPR.
    const _Float16* arow = Ph + (size_t)(u_base + l16) * DIM;
    v8h a0lo = *(const v8h*)(arow +      half * 16);
    v8h a0hi = *(const v8h*)(arow +  8 + half * 16);
    v8h a1lo = *(const v8h*)(arow + 32 + half * 16);
    v8h a1hi = *(const v8h*)(arow + 40 + half * 16);
    v16h a0, a1;
#pragma unroll
    for (int i = 0; i < 8; ++i) {
        a0[i] = a0lo[i];  a0[8 + i] = a0hi[i];
        a1[i] = a1lo[i];  a1[8 + i] = a1hi[i];
    }

    v8f c[4];
#pragma unroll
    for (int t = 0; t < 4; ++t) {
        // ---- B fragment (32x16 per K-block): lane = N (= Q1 row), half-wave
        // splits K 0-15 / 16-31; one contiguous 32B chunk of the Q1 row.
        const _Float16* brow = Q1h + (size_t)(i_base0 + t * 16 + l16) * DIM;
        v16h b0 = *(const v16h*)(brow +      half * 16);
        v16h b1 = *(const v16h*)(brow + 32 + half * 16);

        v8f acc = {};
        acc = __builtin_amdgcn_wmma_f32_16x16x32_f16(
                false, a0, false, b0, (short)0, acc, false, false);
        acc = __builtin_amdgcn_wmma_f32_16x16x32_f16(
                false, a1, false, b1, (short)0, acc, false, false);
        c[t] = acc;
    }

    // ---- epilogue: C/D layout VGPR r -> row r + 8*half, col l16 ----
#pragma unroll
    for (int t = 0; t < 4; ++t) {
        const int col = i_base0 + t * 16 + l16;
        const float bic = bi[col];
#pragma unroll
        for (int r = 0; r < 8; ++r) {
            const int row = u_base + half * 8 + r;
            float x = c[t][r] + bu[row] + bic + U_MEAN;
            float y = 5.0f / (1.0f + __expf(-x));
            __builtin_nontemporal_store(y, out + (size_t)row * n_items + col);
        }
    }
}

// ---------------------------------------------------------------------------
extern "C" void kernel_launch(void* const* d_in, const int* in_sizes, int n_in,
                              void* d_out, int out_size, void* d_ws, size_t ws_size,
                              hipStream_t stream) {
    const float* Q1         = (const float*)d_in[0];
    const float* Q2         = (const float*)d_in[1];
    const float* P          = (const float*)d_in[2];
    const float* bu         = (const float*)d_in[3];
    const float* bi         = (const float*)d_in[4];
    const int*   flat_items = (const int*)d_in[5];
    const int*   seg_ids    = (const int*)d_in[6];

    const int n_users = in_sizes[3];          // 16384
    const int n_items = in_sizes[4];          // 8192
    const int nnz     = in_sizes[5];          // 1048576

    float* out = (float*)d_out;

    // workspace layout: [Ph : U*64 f16][Q1h : I*64 f16]  (~3 MB total)
    _Float16* Ph  = (_Float16*)d_ws;
    _Float16* Q1h = Ph + (size_t)n_users * DIM;

    // 1) convert Q1 to f16
    const int nq = n_items * DIM;
    q1_to_f16_kernel<<<(nq + 255) / 256, 256, 0, stream>>>(Q1, Q1h, nq);

    // 2) ragged segment reduce -> P_full (f16)
    pfull_kernel<<<n_users, 64, 0, stream>>>(Q2, P, flat_items, seg_ids, Ph, nnz);

    // 3) WMMA GEMM + fused sigmoid epilogue
    //    grid.x: n_items/64 N-strips, grid.y: n_users/(16*8) M-groups (8 waves/block)
    dim3 grid(n_items / 64, n_users / (16 * 8));
    gemm_sigmoid_kernel<<<grid, 256, 0, stream>>>(Ph, Q1h, bu, bi, out, n_items);
}
